// CausalSelfAttention_23819888624332
// MI455X (gfx1250) — matrix operand order repaired and checked by reading, not compiled, not run on hardware
//
#include <hip/hip_runtime.h>
#include <stdint.h>

typedef unsigned short u16;
typedef __attribute__((ext_vector_type(16))) __bf16 bf16x16;
typedef __attribute__((ext_vector_type(8)))  float  f32x8;

union AF { uint4 q4[2]; uint32_t u[8]; bf16x16 v; };
union BFr { uint4 q4[2]; uint32_t u[8]; bf16x16 v; };
union CF { float f[8]; f32x8 v; };

__device__ __forceinline__ u16 f2bf(float f) {
  union { float f; uint32_t u; } c; c.f = f;
  uint32_t u = c.u;
  uint32_t r = u + 0x7FFFu + ((u >> 16) & 1u);
  return (u16)(r >> 16);
}

__device__ __forceinline__ f32x8 wmma_bf16(const AF& a, const BFr& b, f32x8 c) {
  return __builtin_amdgcn_wmma_f32_16x16x32_bf16(false, a.v, false, b.v, (short)0, c, false, false);
}

// ---------------- convert kernels ----------------
__global__ __launch_bounds__(256) void conv_f32_bf16(const float* __restrict__ in,
                                                     u16* __restrict__ out, int n) {
  int i = blockIdx.x * 256 + threadIdx.x;
  if (i < n) out[i] = f2bf(in[i]);
}

// in: [rowsK][colsN] f32 row-major; out: [colsN][rowsK] bf16 (i.e. transposed)
__global__ __launch_bounds__(256) void conv_transpose_bf16(const float* __restrict__ in,
                                                           u16* __restrict__ out,
                                                           int rowsK, int colsN) {
  int i = blockIdx.x * 256 + threadIdx.x;
  int total = rowsK * colsN;
  if (i < total) {
    int n = i / rowsK;
    int k = i - n * rowsK;
    out[i] = f2bf(in[(size_t)k * colsN + n]);
  }
}

// ---------------- WMMA GEMM ----------------
// A: bf16 [M][K] row-major.  Bt: bf16 [N][K] (transposed B).
// MODE 0: outF[M][N] fp32 row-major.
// MODE 1: scatter qkv: N=3072 -> q[B,H,T,64], k[B,H,T,64], vT[B,H,64,T] (bf16)
template <int MODE>
__global__ __launch_bounds__(256) void gemm_bf16_kernel(
    const u16* __restrict__ A, const u16* __restrict__ Bt,
    int K, int N, float* __restrict__ outF,
    u16* __restrict__ qO, u16* __restrict__ kO, u16* __restrict__ vTO) {
  const int lane = threadIdx.x & 31;
  const int wave = threadIdx.x >> 5;
  const int half = lane >> 4;
  const int l15  = lane & 15;
  const int m0 = blockIdx.y * 128 + (wave & 3) * 32;
  const int n0 = blockIdx.x * 64  + (wave >> 2) * 32;

  CF acc[2][2];
#pragma unroll
  for (int i = 0; i < 2; ++i)
#pragma unroll
    for (int j = 0; j < 2; ++j)
#pragma unroll
      for (int r = 0; r < 8; ++r) acc[i][j].f[r] = 0.0f;

  const u16* aRow[2] = { A + (size_t)(m0 + l15) * K, A + (size_t)(m0 + 16 + l15) * K };
  const u16* bCol[2] = { Bt + (size_t)(n0 + l15) * K, Bt + (size_t)(n0 + 16 + l15) * K };

  for (int kk = 0; kk < K; kk += 32) {
    AF a[2]; BFr b[2];
#pragma unroll
    for (int i = 0; i < 2; ++i) {
      a[i].q4[0] = *(const uint4*)(aRow[i] + kk + half * 8);        // K 0..15 group
      a[i].q4[1] = *(const uint4*)(aRow[i] + kk + 16 + half * 8);   // K 16..31 group
      b[i].q4[0] = *(const uint4*)(bCol[i] + kk + half * 8);
      b[i].q4[1] = *(const uint4*)(bCol[i] + kk + 16 + half * 8);
    }
    acc[0][0].v = wmma_bf16(a[0], b[0], acc[0][0].v);
    acc[0][1].v = wmma_bf16(a[0], b[1], acc[0][1].v);
    acc[1][0].v = wmma_bf16(a[1], b[0], acc[1][0].v);
    acc[1][1].v = wmma_bf16(a[1], b[1], acc[1][1].v);
  }

#pragma unroll
  for (int i = 0; i < 2; ++i)
#pragma unroll
    for (int j = 0; j < 2; ++j) {
      const int colN = n0 + j * 16 + l15;
#pragma unroll
      for (int r = 0; r < 8; ++r) {
        const int rowM = m0 + i * 16 + half * 8 + r;
        const float val = acc[i][j].f[r];
        if (MODE == 0) {
          outF[(size_t)rowM * N + colN] = val;
        } else {
          const int bb = rowM >> 11, t = rowM & 2047;
          const int sec = colN >> 10, c = colN & 1023;
          const int h = c >> 6, d = c & 63;
          const int bh = bb * 16 + h;
          const u16 bv = f2bf(val);
          if (sec == 0)      qO[((size_t)bh * 2048 + t) * 64 + d] = bv;
          else if (sec == 1) kO[((size_t)bh * 2048 + t) * 64 + d] = bv;
          else               vTO[((size_t)bh * 64 + d) * 2048 + t] = bv;
        }
      }
    }
}

// ---------------- flash attention (one wave per 16-query tile) ----------------
__global__ __launch_bounds__(256) void attn_kernel(const u16* __restrict__ q,
                                                   const u16* __restrict__ k,
                                                   const u16* __restrict__ vT,
                                                   u16* __restrict__ aout) {
  const int lane = threadIdx.x & 31;
  const int wave = threadIdx.x >> 5;
  const int half = lane >> 4;
  const int l15  = lane & 15;

  const int idx = blockIdx.x * 8 + wave;   // 0..8191
  const int qt  = idx & 127;               // 16-row query tile
  const int bh  = idx >> 7;                // b*16 + h

  const u16* qb = q  + (size_t)bh * 2048 * 64;
  const u16* kb = k  + (size_t)bh * 2048 * 64;
  const u16* vb = vT + (size_t)bh * 64 * 2048;

  __shared__ u16 lds_p[8][512];            // per-wave 32x16 bf16 P tile (col-major)
  u16* myp = lds_p[wave];

  // Q fragments: two K=32 chunks over D=64
  AF qa[2];
  {
    const int trow = qt * 16 + l15;
#pragma unroll
    for (int c = 0; c < 2; ++c) {
      qa[c].q4[0] = *(const uint4*)(qb + (size_t)trow * 64 + c * 32 + half * 8);
      qa[c].q4[1] = *(const uint4*)(qb + (size_t)trow * 64 + c * 32 + 16 + half * 8);
    }
  }

  CF o[4];
#pragma unroll
  for (int t4 = 0; t4 < 4; ++t4)
#pragma unroll
    for (int r = 0; r < 8; ++r) o[t4].f[r] = 0.0f;
  float mrow[8], lrow[8];
#pragma unroll
  for (int r = 0; r < 8; ++r) { mrow[r] = -__builtin_inff(); lrow[r] = 0.0f; }

  const int nchunk = (qt >> 1) + 1;        // 32-key chunks to cover causal range
  for (int ch = 0; ch < nchunk; ++ch) {
    const int key0 = ch * 32;
    CF s[2];
#pragma unroll
    for (int sub = 0; sub < 2; ++sub) {
      const int kt = key0 + sub * 16;
      BFr kf0, kf1;
      const u16* krow = kb + (size_t)(kt + l15) * 64;
      kf0.q4[0] = *(const uint4*)(krow + half * 8);
      kf0.q4[1] = *(const uint4*)(krow + 16 + half * 8);
      kf1.q4[0] = *(const uint4*)(krow + 32 + half * 8);
      kf1.q4[1] = *(const uint4*)(krow + 32 + 16 + half * 8);
      CF z;
#pragma unroll
      for (int r = 0; r < 8; ++r) z.f[r] = 0.0f;
      s[sub].v = wmma_bf16(qa[0], kf0, z.v);
      s[sub].v = wmma_bf16(qa[1], kf1, s[sub].v);
      const int kcol = kt + l15;
#pragma unroll
      for (int r = 0; r < 8; ++r) {
        const int qrow = qt * 16 + half * 8 + r;
        const float val = s[sub].f[r] * 0.125f;  // 1/sqrt(64)
        s[sub].f[r] = (kcol <= qrow) ? val : -__builtin_inff();
      }
    }
    // online softmax (rows live on fixed lane-halves; butterfly over 16 lanes)
    float alpha[8];
#pragma unroll
    for (int r = 0; r < 8; ++r) {
      float mx = fmaxf(s[0].f[r], s[1].f[r]);
#pragma unroll
      for (int m = 1; m < 16; m <<= 1) mx = fmaxf(mx, __shfl_xor(mx, m, 32));
      const float mnew = fmaxf(mrow[r], mx);
      const float al = __expf(mrow[r] - mnew);
      const float p0 = __expf(s[0].f[r] - mnew);
      const float p1 = __expf(s[1].f[r] - mnew);
      s[0].f[r] = p0; s[1].f[r] = p1;
      float rs = p0 + p1;
#pragma unroll
      for (int m = 1; m < 16; m <<= 1) rs += __shfl_xor(rs, m, 32);
      lrow[r] = lrow[r] * al + rs;
      mrow[r] = mnew;
      alpha[r] = al;
    }
#pragma unroll
    for (int t4 = 0; t4 < 4; ++t4)
#pragma unroll
      for (int r = 0; r < 8; ++r) o[t4].f[r] *= alpha[r];

    // P -> LDS (col-major 32x16), then reload as A-fragment
#pragma unroll
    for (int sub = 0; sub < 2; ++sub) {
      uint32_t pw[4];
#pragma unroll
      for (int i = 0; i < 4; ++i)
        pw[i] = (uint32_t)f2bf(s[sub].f[2 * i]) | ((uint32_t)f2bf(s[sub].f[2 * i + 1]) << 16);
      const int col = sub * 16 + l15;
      *(uint4*)(myp + col * 16 + half * 8) = make_uint4(pw[0], pw[1], pw[2], pw[3]);
    }
    asm volatile("s_wait_dscnt 0x0" ::: "memory");
    AF pa;
#pragma unroll
    for (int v = 0; v < 8; ++v) {
      const int kidx = (v >> 2) * 16 + half * 8 + (v & 3) * 2;
      const uint32_t lo = myp[kidx * 16 + l15];
      const uint32_t hi = myp[(kidx + 1) * 16 + l15];
      pa.u[v] = lo | (hi << 16);
    }
    // P @ V : 4 WMMAs over D chunks of 16
#pragma unroll
    for (int t4 = 0; t4 < 4; ++t4) {
      BFr vf;
      const u16* vrow = vb + (size_t)(t4 * 16 + l15) * 2048 + key0;
      vf.q4[0] = *(const uint4*)(vrow + half * 8);
      vf.q4[1] = *(const uint4*)(vrow + 16 + half * 8);
      o[t4].v = wmma_bf16(pa, vf, o[t4].v);
    }
  }

  // normalize + write bf16 attention output [B*T][C]
  float inv[8];
#pragma unroll
  for (int r = 0; r < 8; ++r) inv[r] = 1.0f / lrow[r];
  const int b = bh >> 4, h = bh & 15;
#pragma unroll
  for (int t4 = 0; t4 < 4; ++t4) {
    const int acol = h * 64 + t4 * 16 + l15;
#pragma unroll
    for (int r = 0; r < 8; ++r) {
      const int arow = b * 2048 + qt * 16 + half * 8 + r;
      aout[(size_t)arow * 1024 + acol] = f2bf(o[t4].f[r] * inv[r]);
    }
  }
}

// ---------------- launcher ----------------
extern "C" void kernel_launch(void* const* d_in, const int* in_sizes, int n_in,
                              void* d_out, int out_size, void* d_ws, size_t ws_size,
                              hipStream_t stream) {
  (void)in_sizes; (void)n_in; (void)out_size; (void)ws_size;
  const float* x      = (const float*)d_in[0];   // [4,2048,1024]
  const float* w_attn = (const float*)d_in[1];   // [1024,3072]
  const float* w_proj = (const float*)d_in[2];   // [1024,1024]
  float* out = (float*)d_out;                    // [4,2048,1024]

  const int M = 4 * 2048;       // 8192
  const int C = 1024;
  const int N3 = 3 * C;         // 3072
  const size_t HD = (size_t)4 * 16 * 2048 * 64; // 8388608 elems per q/k/v

  u16* ws = (u16*)d_ws;
  size_t off = 0;
  u16* xb     = ws + off; off += (size_t)M * C;      // x bf16
  u16* wattnT = ws + off; off += (size_t)N3 * C;     // w_attn^T bf16
  u16* wprojT = ws + off; off += (size_t)C * C;      // w_proj^T bf16
  u16* qB     = ws + off; off += HD;                 // q [B,H,T,64]
  u16* kB     = ws + off; off += HD;                 // k [B,H,T,64]
  u16* vTB    = ws + off; off += HD;                 // v^T [B,H,64,T]
  u16* aoutB  = ws + off; off += (size_t)M * C;      // attn out bf16

  {
    int n = M * C;
    conv_f32_bf16<<<(n + 255) / 256, 256, 0, stream>>>(x, xb, n);
  }
  {
    int n = C * N3;
    conv_transpose_bf16<<<(n + 255) / 256, 256, 0, stream>>>(w_attn, wattnT, C, N3);
  }
  {
    int n = C * C;
    conv_transpose_bf16<<<(n + 255) / 256, 256, 0, stream>>>(w_proj, wprojT, C, C);
  }

  gemm_bf16_kernel<1><<<dim3(N3 / 64, M / 128), 256, 0, stream>>>(
      xb, wattnT, C, N3, nullptr, qB, kB, vTB);

  attn_kernel<<<1024, 256, 0, stream>>>(qB, kB, vTB, aoutB);

  gemm_bf16_kernel<0><<<dim3(C / 64, M / 128), 256, 0, stream>>>(
      aoutB, wprojT, C, C, out, nullptr, nullptr, nullptr);
}
